// BayesLinear_63247688401633
// MI455X (gfx1250) — matrix operand order repaired and checked by reading, not compiled, not run on hardware
//
#include <hip/hip_runtime.h>

#define NROWS 8192
#define KDIM  4096
#define ODIM  16384

typedef __bf16 bf16_t;
typedef __attribute__((ext_vector_type(16))) __bf16 bf16x16;
typedef __attribute__((ext_vector_type(4)))  __bf16 bf16x4;
typedef __attribute__((ext_vector_type(8)))  float  f32x8;

// ---------------------------------------------------------------------------
// Prep kernels: fuse reparameterization + fp32->bf16 downconvert (one pass).
// ---------------------------------------------------------------------------

__global__ __launch_bounds__(256) void bayes_cvt_weight(
    const float* __restrict__ mu, const float* __restrict__ ls,
    const float* __restrict__ eps, bf16_t* __restrict__ out)
{
    size_t i = ((size_t)blockIdx.x * 256 + threadIdx.x) * 4;
    float4 m = *(const float4*)(mu + i);
    float4 s = *(const float4*)(ls + i);
    float4 e = *(const float4*)(eps + i);
    bf16x4 r;
    r[0] = (bf16_t)(m.x + __expf(s.x) * e.x);
    r[1] = (bf16_t)(m.y + __expf(s.y) * e.y);
    r[2] = (bf16_t)(m.z + __expf(s.z) * e.z);
    r[3] = (bf16_t)(m.w + __expf(s.w) * e.w);
    *(bf16x4*)(out + i) = r;
}

__global__ __launch_bounds__(256) void bayes_cvt_x(
    const float* __restrict__ x, bf16_t* __restrict__ out)
{
    size_t i = ((size_t)blockIdx.x * 256 + threadIdx.x) * 4;
    float4 v = *(const float4*)(x + i);
    bf16x4 r;
    r[0] = (bf16_t)v.x; r[1] = (bf16_t)v.y; r[2] = (bf16_t)v.z; r[3] = (bf16_t)v.w;
    *(bf16x4*)(out + i) = r;
}

__global__ __launch_bounds__(256) void bayes_cvt_bias(
    const float* __restrict__ mu, const float* __restrict__ ls,
    const float* __restrict__ eps, float* __restrict__ out)
{
    int i = blockIdx.x * 256 + threadIdx.x;
    out[i] = mu[i] + __expf(ls[i]) * eps[i];
}

// ---------------------------------------------------------------------------
// bf16 WMMA GEMM:  C[n,o] = sum_k xb[n,k] * wb[o,k] + bias[o]
// Block = 256 threads (8 waves), 128x256 output tile, K-step = 32.
// Wave grid 2x4: each wave -> 64x64 = 4x4 tiles of 16x16 (16 WMMAs/K-step).
// Tile feed: GLOBAL_LOAD_ASYNC_TO_LDS_B128 double-buffered, ASYNCcnt-paced.
// ---------------------------------------------------------------------------

__global__ __launch_bounds__(256) void bayes_gemm_bf16(
    const bf16_t* __restrict__ A,    // [NROWS][KDIM]  x (bf16)
    const bf16_t* __restrict__ B,    // [ODIM][KDIM]   W (bf16)
    const float*  __restrict__ bias, // [ODIM]
    float*        __restrict__ C)    // [NROWS][ODIM]
{
    constexpr int LDK = 40;                 // padded row (bank-spread), 16B-aligned rows
    __shared__ bf16_t As[2][128][LDK];      // 2 x 10 KB
    __shared__ bf16_t Bs[2][256][LDK];      // 2 x 20 KB

    const int tid  = threadIdx.x;
    const int lane = tid & 31;
    const int wave = tid >> 5;
    const int lm   = lane & 15;             // M (A) / N (B) index within 16x16 tile
    const int lh   = lane >> 4;             // lane-half selects K chunk

    const int waveCol = wave & 3;           // 0..3 -> 64-col strip
    const int waveRow = wave >> 2;          // 0..1 -> 64-row strip

    const int col0 = blockIdx.x * 256;      // OUT dimension
    const int row0 = blockIdx.y * 128;      // N dimension

    // ---- Async staging map: 384 tile rows x 4 16B-chunks = 1536 chunks,
    //      6 chunks per thread. Rows 0..127 -> A tile, 128..383 -> B tile.
    const bf16_t* gsrc[6];
    unsigned ldst[2][6];                    // LDS byte offsets (low 32 bits of generic ptr)
    #pragma unroll
    for (int j = 0; j < 6; ++j) {
        const int chunk = tid + 256 * j;
        const int row   = chunk >> 2;       // 0..383
        const int kq    = (chunk & 3) * 8;  // element offset: 0,8,16,24
        if (row < 128) {
            gsrc[j]    = A + (size_t)(row0 + row) * KDIM + kq;
            ldst[0][j] = (unsigned)(unsigned long long)&As[0][row][kq];
            ldst[1][j] = (unsigned)(unsigned long long)&As[1][row][kq];
        } else {
            const int c = row - 128;
            gsrc[j]    = B + (size_t)(col0 + c) * KDIM + kq;
            ldst[0][j] = (unsigned)(unsigned long long)&Bs[0][c][kq];
            ldst[1][j] = (unsigned)(unsigned long long)&Bs[1][c][kq];
        }
    }

    f32x8 acc[4][4] = {};

    // Issue tile 0 into buffer 0.
    #pragma unroll
    for (int j = 0; j < 6; ++j)
        asm volatile("global_load_async_to_lds_b128 %0, %1, off"
                     :: "v"(ldst[0][j]), "v"((unsigned long long)gsrc[j])
                     : "memory");

    const int NKT = KDIM / 32;              // 128 K-steps
    for (int kt = 0; kt < NKT; ++kt) {
        const int buf = kt & 1;

        if (kt + 1 < NKT) {
            // Issue next tile into the other buffer (reads of it finished at the
            // trailing barrier of the previous iteration).
            #pragma unroll
            for (int j = 0; j < 6; ++j)
                asm volatile("global_load_async_to_lds_b128 %0, %1, off"
                             :: "v"(ldst[buf ^ 1][j]),
                                "v"((unsigned long long)(gsrc[j] + (size_t)(kt + 1) * 32))
                             : "memory");
            // In-order completion: <=6 outstanding means tile-kt's 6 are done.
            asm volatile("s_wait_asynccnt 6" ::: "memory");
        } else {
            asm volatile("s_wait_asynccnt 0" ::: "memory");
        }
        __syncthreads();                    // tile kt visible to all waves

        // Fragments per ISA VGPR layouts.
        union FragU { bf16x16 v; uint4 q[2]; } fa[4], fb[4];
        const int kA = lh * 8;              // A: lanes 0-15 -> K{0..7,16..23}; 16-31 -> K{8..15,24..31}
        const int kB = lh * 8;              // B: lanes 0-15 -> K{0..7,16..23}; 16-31 -> K{8..15,24..31}
        #pragma unroll
        for (int m = 0; m < 4; ++m) {
            const int r = waveRow * 64 + m * 16 + lm;
            fa[m].q[0] = *(const uint4*)&As[buf][r][kA];
            fa[m].q[1] = *(const uint4*)&As[buf][r][kA + 16];
        }
        #pragma unroll
        for (int n = 0; n < 4; ++n) {
            const int c = waveCol * 64 + n * 16 + lm;
            fb[n].q[0] = *(const uint4*)&Bs[buf][c][kB];
            fb[n].q[1] = *(const uint4*)&Bs[buf][c][kB + 16];
        }

        // 16 WMMAs per wave per K-step.
        #pragma unroll
        for (int m = 0; m < 4; ++m)
            #pragma unroll
            for (int n = 0; n < 4; ++n)
                acc[m][n] = __builtin_amdgcn_wmma_f32_16x16x32_bf16(
                    false, fa[m].v, false, fb[n].v,
                    (short)0, acc[m][n], false, false);

        __syncthreads();                    // reads done before this buf is refilled
    }

    // Epilogue: D layout -> M = vgpr + 8*(lane>=16), N = lane%16. Fuse bias.
    #pragma unroll
    for (int n = 0; n < 4; ++n) {
        const int c  = col0 + waveCol * 64 + n * 16 + lm;
        const float bv = bias[c];
        #pragma unroll
        for (int m = 0; m < 4; ++m) {
            const int rbase = row0 + waveRow * 64 + m * 16 + lh * 8;
            #pragma unroll
            for (int v = 0; v < 8; ++v)
                C[(size_t)(rbase + v) * ODIM + c] = acc[m][n][v] + bv;
        }
    }
}

// ---------------------------------------------------------------------------
// Launcher
// ---------------------------------------------------------------------------

extern "C" void kernel_launch(void* const* d_in, const int* in_sizes, int n_in,
                              void* d_out, int out_size, void* d_ws, size_t ws_size,
                              hipStream_t stream) {
    const float* x   = (const float*)d_in[0];
    const float* wmu = (const float*)d_in[1];
    const float* wls = (const float*)d_in[2];
    const float* bmu = (const float*)d_in[3];
    const float* bls = (const float*)d_in[4];
    const float* ew  = (const float*)d_in[5];
    const float* eb  = (const float*)d_in[6];
    float* out = (float*)d_out;

    // Workspace layout: x_bf16 (64 MiB) | W_bf16 (128 MiB) | bias_f32 (64 KiB)
    bf16_t* xb = (bf16_t*)d_ws;
    bf16_t* wb = (bf16_t*)((char*)d_ws + (size_t)NROWS * KDIM * sizeof(bf16_t));
    float* biasf = (float*)((char*)d_ws + (size_t)NROWS * KDIM * sizeof(bf16_t)
                                        + (size_t)ODIM  * KDIM * sizeof(bf16_t));

    // Prep: reparameterize + downconvert (single bandwidth-bound pass).
    bayes_cvt_weight<<<(ODIM * (size_t)KDIM) / (4 * 256), 256, 0, stream>>>(wmu, wls, ew, wb);
    bayes_cvt_x     <<<(NROWS * (size_t)KDIM) / (4 * 256), 256, 0, stream>>>(x, xb);
    bayes_cvt_bias  <<<ODIM / 256, 256, 0, stream>>>(bmu, bls, eb, biasf);

    // GEMM: grid over (OUT/256, N/128) output tiles.
    dim3 grid(ODIM / 256, NROWS / 128);
    bayes_gemm_bf16<<<grid, 256, 0, stream>>>(xb, wb, biasf, out);
}
